// EgretPPI_22814866276779
// MI455X (gfx1250) — compile-verified, matrix operand order checked by reading, not run on hardware
//
#include <hip/hip_runtime.h>

typedef __attribute__((ext_vector_type(16))) _Float16 v16h;
typedef __attribute__((ext_vector_type(8)))  _Float16 v8h;
typedef __attribute__((ext_vector_type(4)))  _Float16 v4h;
typedef __attribute__((ext_vector_type(8)))  float    v8f;

namespace {
constexpr int B_  = 32;
constexpr int L_  = 1024;
constexpr int H_  = 4;
constexpr int NL_ = 2;
constexpr int DB_ = 1024;
constexpr int DP_ = 7;
constexpr int DA_ = 2;
constexpr int CIN_ = DB_ + DP_ + DA_;   // 1033
constexpr int D_  = 32;
constexpr int N_  = B_ * L_;            // 32768
constexpr int K_  = 20;
constexpr int E_  = N_ * K_;            // 655360
constexpr int KCH_ = 33;                // ceil(1033/32)

constexpr int ROWS_ = 134;              // 128 positions + 6 halo
constexpr int ASTR_ = 40;               // LDS A row stride (halfs): 80B, 16B-aligned, conflict-free
constexpr int BCHUNK_ = 7 * 2 * 32 * 16; // halfs of B frags per k-chunk = 7168

// workspace layout (bytes)
constexpr size_t WB_OFF   = 0;                                // conv W frags: 33*7*2*32*16 halfs = 473088 B
constexpr size_t FCW_OFF  = 473088;                           // fc_w frags: 2*4*2*32*16 halfs = 16384 B
constexpr size_t COEF_OFF = FCW_OFF + 16384;                  // 64 B (pad 256)
constexpr size_t FEATS_OFF= COEF_OFF + 256;                   // N*32 f32 = 4 MiB
constexpr size_t H16_OFF  = FEATS_OFF + (size_t)N_*32*4;      // N*32 f16 = 2 MiB
constexpr size_t Z_OFF    = H16_OFF  + (size_t)N_*32*2;       // N*H*32 f32 = 16 MiB
constexpr size_t SSRC_OFF = Z_OFF    + (size_t)N_*H_*32*4;
constexpr size_t SDST_OFF = SSRC_OFF + (size_t)N_*H_*4;
constexpr size_t M_OFF    = SDST_OFF + (size_t)N_*H_*4;
constexpr size_t SUM_OFF  = M_OFF    + (size_t)N_*H_*4;
constexpr size_t EB_OFF   = SUM_OFF  + (size_t)N_*H_*4;       // E*H f32 = 10 MiB
constexpr size_t HACC_OFF = EB_OFF   + (size_t)E_*H_*4;       // N*32 f32 = 4 MiB
} // namespace

// monotone float<->uint mapping for atomic max on floats
__device__ __forceinline__ unsigned fmap(float x) {
  unsigned u = __float_as_uint(x);
  return (u & 0x80000000u) ? ~u : (u | 0x80000000u);
}
__device__ __forceinline__ float funmap(unsigned u) {
  unsigned b = (u & 0x80000000u) ? (u ^ 0x80000000u) : ~u;
  return __uint_as_float(b);
}

__device__ __forceinline__ float load_chan(const float* __restrict__ pb,
                                           const float* __restrict__ pc,
                                           const float* __restrict__ ra,
                                           int b, int t, int c) {
  if (c < DB_) return pb[((size_t)b * L_ + t) * DB_ + c];
  c -= DB_;
  if (c < DP_) return pc[((size_t)b * L_ + t) * DP_ + c];
  c -= DP_;
  if (c < DA_) return ra[((size_t)b * L_ + t) * DA_ + c];
  return 0.0f;
}

// ---------------- prep: pack conv weights into WMMA B-fragment layout -------
// kc-major so per-chunk staging copy is contiguous:
// frag[(((kc*7 + tap)*2 + ct)*32 + lane)*16 + j]
__global__ void prep_convw_kernel(const float* __restrict__ conv_w,
                                  _Float16* __restrict__ wB) {
  int tid = blockIdx.x * blockDim.x + threadIdx.x;
  if (tid >= KCH_ * 7 * 2 * 32) return;
  int lane = tid & 31;
  int ct   = (tid >> 5) & 1;
  int rest = tid >> 6;
  int tap  = rest % 7;
  int kc   = rest / 7;
  int o    = ct * 16 + (lane & 15);
  int klo  = (lane & 16) ? 8 : 0;
  _Float16* dst = wB + ((size_t)tid) * 16;
  #pragma unroll
  for (int j = 0; j < 8; ++j) {
    int c0 = kc * 32 + klo + j;
    int c1 = c0 + 16;
    float v0 = (c0 < CIN_) ? conv_w[((size_t)o * CIN_ + c0) * 7 + tap] : 0.0f;
    float v1 = (c1 < CIN_) ? conv_w[((size_t)o * CIN_ + c1) * 7 + tap] : 0.0f;
    dst[j]     = (_Float16)v0;
    dst[8 + j] = (_Float16)v1;
  }
}

// pack fc_w (NL,H,D,D) into B-fragments for z = h * fc_w^T
__global__ void prep_fcw_kernel(const float* __restrict__ fc_w,
                                _Float16* __restrict__ fw) {
  int tid = blockIdx.x * blockDim.x + threadIdx.x;
  if (tid >= NL_ * H_ * 2 * 32) return;
  int lane = tid & 31;
  int ct   = (tid >> 5) & 1;
  int h    = (tid >> 6) & 3;
  int l    = tid >> 8;
  int o    = ct * 16 + (lane & 15);
  int klo  = (lane & 16) ? 8 : 0;
  _Float16* dst = fw + ((size_t)tid) * 16;
  #pragma unroll
  for (int j = 0; j < 8; ++j) {
    int d0 = klo + j;
    int d1 = d0 + 16;
    dst[j]     = (_Float16)fc_w[(((size_t)(l * H_ + h)) * D_ + o) * D_ + d0];
    dst[8 + j] = (_Float16)fc_w[(((size_t)(l * H_ + h)) * D_ + o) * D_ + d1];
  }
}

// per (l,h) edge-attention coefficients: e += ca*f0 + cb*f1
__global__ void prep_coef_kernel(const float* __restrict__ attn_w,
                                 const float* __restrict__ fc_eatt_w,
                                 float* __restrict__ coef) {
  int tid = blockIdx.x * blockDim.x + threadIdx.x;
  if (tid >= NL_ * H_) return;
  int l = tid / H_, h = tid % H_;
  const float* ae = attn_w + (size_t)(l * H_ + h) * (2 * D_ + 2) + 2 * D_;
  const float* w  = fc_eatt_w + (size_t)(l * H_ + h) * 4;  // [g][f]
  coef[tid * 2 + 0] = ae[0] * w[0] + ae[1] * w[2];
  coef[tid * 2 + 1] = ae[0] * w[1] + ae[1] * w[3];
}

// ---------------- encoder: conv1d(k=7,pad=3) + leaky + BN ------------------
// Block = 128 positions (8 waves x 16). k-chunk outer / tap inner with
// double-buffered LDS staging of A (halo rows, f32->f16 once) and B (packed
// weight fragments), so global-load latency hides behind the WMMAs.
__global__ __launch_bounds__(256) void conv_encoder_kernel(
    const float* __restrict__ pb, const float* __restrict__ pc,
    const float* __restrict__ ra, const _Float16* __restrict__ wB,
    const float* __restrict__ cb, const float* __restrict__ gamma,
    const float* __restrict__ beta, const float* __restrict__ mean,
    const float* __restrict__ var, float* __restrict__ feats,
    _Float16* __restrict__ h16) {
  __shared__ __align__(32) _Float16 ldsA[2][ROWS_ * ASTR_];
  __shared__ __align__(32) _Float16 ldsB[2][BCHUNK_];

  int tid  = threadIdx.x;
  int wave = tid >> 5;
  int lane = tid & 31;
  int pB   = blockIdx.x * 128;     // block position base (never crosses batch)
  int bb   = pB >> 10;
  int tB   = pB & 1023;
  int klo  = (lane & 16) ? 8 : 0;

  auto stageA = [&](int kc, int s) {
    // 134 rows x 8 quad-channels; zero-pad outside [0,L)
    for (int i = tid; i < ROWS_ * 8; i += 256) {
      int r  = i >> 3;
      int c4 = (i & 7) * 4;
      int t  = tB - 3 + r;
      float x0 = 0.f, x1 = 0.f, x2 = 0.f, x3 = 0.f;
      if (t >= 0 && t < L_) {
        if (kc < 32) {
          float4 v = *(const float4*)(pb + ((size_t)bb * L_ + t) * DB_ + kc * 32 + c4);
          x0 = v.x; x1 = v.y; x2 = v.z; x3 = v.w;
        } else {
          int c = kc * 32 + c4;
          x0 = load_chan(pb, pc, ra, bb, t, c);
          x1 = load_chan(pb, pc, ra, bb, t, c + 1);
          x2 = load_chan(pb, pc, ra, bb, t, c + 2);
          x3 = load_chan(pb, pc, ra, bb, t, c + 3);
        }
      }
      v4h h4;
      h4[0] = (_Float16)x0; h4[1] = (_Float16)x1;
      h4[2] = (_Float16)x2; h4[3] = (_Float16)x3;
      *(v4h*)&ldsA[s][r * ASTR_ + c4] = h4;
    }
  };
  auto stageB = [&](int kc, int s) {
    const float4* src = (const float4*)(wB + (size_t)kc * BCHUNK_);
    float4* dst = (float4*)&ldsB[s][0];
    for (int i = tid; i < BCHUNK_ / 8; i += 256) dst[i] = src[i];
  };

  v8f acc0 = {};
  v8f acc1 = {};
  stageA(0, 0);
  stageB(0, 0);
  for (int kc = 0; kc < KCH_; ++kc) {
    int s = kc & 1;
    __syncthreads();                       // buf s staged; prev reads of s^1 done
    if (kc + 1 < KCH_) { stageA(kc + 1, s ^ 1); stageB(kc + 1, s ^ 1); }
    #pragma unroll
    for (int tap = 0; tap < 7; ++tap) {
      int rr = wave * 16 + (lane & 15) + tap;   // LDS halo row for this lane
      v16h a;
      {
        v8h lo = *(const v8h*)&ldsA[s][rr * ASTR_ + klo];
        v8h hi = *(const v8h*)&ldsA[s][rr * ASTR_ + klo + 16];
        #pragma unroll
        for (int j = 0; j < 8; ++j) { a[j] = lo[j]; a[8 + j] = hi[j]; }
      }
      v16h b0 = *(const v16h*)&ldsB[s][(size_t)((tap * 2 + 0) * 32 + lane) * 16];
      v16h b1 = *(const v16h*)&ldsB[s][(size_t)((tap * 2 + 1) * 32 + lane) * 16];
      acc0 = __builtin_amdgcn_wmma_f32_16x16x32_f16(false, a, false, b0,
                                                    (short)0, acc0, false, false);
      acc1 = __builtin_amdgcn_wmma_f32_16x16x32_f16(false, a, false, b1,
                                                    (short)0, acc1, false, false);
    }
  }

  // epilogue: bias + leaky + BN(eval); D layout: lane -> col, vgpr r -> row
  int col = lane & 15;
  int rbase = (lane >> 4) * 8;
  int p0 = pB + wave * 16;
  #pragma unroll
  for (int ct = 0; ct < 2; ++ct) {
    int o = ct * 16 + col;
    float bias = cb[o];
    float sc = gamma[o] * rsqrtf(var[o] + 1e-5f);
    float mu = mean[o], be = beta[o];
    v8f acc = ct ? acc1 : acc0;
    #pragma unroll
    for (int r = 0; r < 8; ++r) {
      int p = p0 + rbase + r;
      float y = acc[r] + bias;
      y = (y >= 0.0f) ? y : 0.01f * y;
      y = (y - mu) * sc + be;
      feats[(size_t)p * 32 + o] = y;
      h16[(size_t)p * 32 + o] = (_Float16)y;
    }
  }
}

// ---------------- GNN: z = h * fc_w^T per head (WMMA) ----------------------
__global__ __launch_bounds__(256) void z_kernel(const _Float16* __restrict__ h16,
                                                const _Float16* __restrict__ fw,
                                                float* __restrict__ z, int l) {
  int wave = threadIdx.x >> 5;
  int lane = threadIdx.x & 31;
  int tile = blockIdx.x * 8 + wave;
  int n0 = tile * 16;
  int row = lane & 15;
  int klo = (lane & 16) ? 8 : 0;
  int n = n0 + row;
  v16h a;
  {
    v8h lo = *(const v8h*)(h16 + (size_t)n * 32 + klo);
    v8h hi = *(const v8h*)(h16 + (size_t)n * 32 + klo + 16);
    #pragma unroll
    for (int j = 0; j < 8; ++j) { a[j] = lo[j]; a[8 + j] = hi[j]; }
  }
  int col = lane & 15;
  int rbase = (lane >> 4) * 8;
  for (int h = 0; h < H_; ++h) {
    #pragma unroll
    for (int ct = 0; ct < 2; ++ct) {
      const _Float16* wp = fw + ((size_t)(((l * H_ + h) * 2 + ct) * 32) + lane) * 16;
      v16h b = *(const v16h*)wp;
      v8f acc = {};
      acc = __builtin_amdgcn_wmma_f32_16x16x32_f16(false, a, false, b,
                                                   (short)0, acc, false, false);
      int o = ct * 16 + col;
      #pragma unroll
      for (int r = 0; r < 8; ++r) {
        int nn = n0 + rbase + r;
        z[((size_t)nn * H_ + h) * 32 + o] = acc[r];
      }
    }
  }
}

// node partial scores + zero the per-layer reduction buffers
__global__ void s_init_kernel(const float* __restrict__ z,
                              const float* __restrict__ attn_w,
                              float* __restrict__ ssrc, float* __restrict__ sdst,
                              unsigned* __restrict__ m, float* __restrict__ sum,
                              float* __restrict__ hacc, int l) {
  int idx = blockIdx.x * blockDim.x + threadIdx.x;   // N*H
  if (idx >= N_ * H_) return;
  int n = idx >> 2, h = idx & 3;
  const float* aw = attn_w + (size_t)(l * H_ + h) * (2 * D_ + 2);
  const float* zp = z + (size_t)idx * 32;
  float s0 = 0.0f, s1 = 0.0f;
  #pragma unroll
  for (int d = 0; d < 32; ++d) { s0 += zp[d] * aw[d]; s1 += zp[d] * aw[32 + d]; }
  ssrc[idx] = s0;
  sdst[idx] = s1;
  m[idx] = 0u;          // < fmap of any finite value: safe identity for mapped max
  sum[idx] = 0.0f;
  float* hp = hacc + (size_t)n * 32 + h * 8;
  #pragma unroll
  for (int j = 0; j < 8; ++j) hp[j] = 0.0f;
}

// pass 1: raw attention logits + segment max (mapped-uint atomicMax)
__global__ void e1_kernel(const float* __restrict__ ef, const int* __restrict__ si,
                          const int* __restrict__ di, const float* __restrict__ ssrc,
                          const float* __restrict__ sdst, const float* __restrict__ coef,
                          float* __restrict__ ebuf, unsigned* __restrict__ m, int l) {
  int i = blockIdx.x * blockDim.x + threadIdx.x;
  if (i >= E_) return;
  float f0 = ef[2 * i], f1 = ef[2 * i + 1];
  int s = si[i], d = di[i];
  #pragma unroll
  for (int h = 0; h < H_; ++h) {
    float ca = coef[(l * H_ + h) * 2], cbv = coef[(l * H_ + h) * 2 + 1];
    float e = ssrc[s * H_ + h] + sdst[d * H_ + h] + ca * f0 + cbv * f1;
    e = (e >= 0.0f) ? e : 0.01f * e;
    ebuf[(size_t)i * H_ + h] = e;
    atomicMax(&m[d * H_ + h], fmap(e));
  }
}

// pass 2: exp(e - max) + segment sum
__global__ void e2_kernel(const int* __restrict__ di, float* __restrict__ ebuf,
                          const unsigned* __restrict__ m, float* __restrict__ sum) {
  int idx = blockIdx.x * blockDim.x + threadIdx.x;   // E*H
  if (idx >= E_ * H_) return;
  int i = idx >> 2, h = idx & 3;
  int d = di[i];
  float ex = __expf(ebuf[idx] - funmap(m[d * H_ + h]));
  ebuf[idx] = ex;
  atomicAdd(&sum[d * H_ + h], ex);
}

// pass 3: alpha = ex/sum; scatter alpha*(z[src] + ez) into hacc
__global__ void e3_kernel(const float* __restrict__ ef, const int* __restrict__ si,
                          const int* __restrict__ di, const float* __restrict__ ebuf,
                          const float* __restrict__ sum, const float* __restrict__ z,
                          const float* __restrict__ few, float* __restrict__ hacc,
                          float* __restrict__ alpha_out, int l, int last) {
  int idx = blockIdx.x * blockDim.x + threadIdx.x;   // E*H
  if (idx >= E_ * H_) return;
  int i = idx >> 2, h = idx & 3;
  int s = si[i], d = di[i];
  float al = ebuf[idx] / sum[d * H_ + h];
  if (last) alpha_out[idx] = al;
  float f0 = ef[2 * i], f1 = ef[2 * i + 1];
  const float* zp = z + ((size_t)s * H_ + h) * 32;
  const float* ew = few + (size_t)(l * H_ + h) * D_ * 2;  // (NL,H,D,2)
  float* hp = hacc + (size_t)d * 32;
  #pragma unroll
  for (int o = 0; o < 32; ++o) {
    float v = zp[o] + ew[2 * o] * f0 + ew[2 * o + 1] * f1;
    atomicAdd(&hp[o], al * v);
  }
}

// head-mean epilogue; on last layer write concat(h_final, feats) output
__global__ void n2_kernel(const float* __restrict__ hacc, const float* __restrict__ feats,
                          _Float16* __restrict__ h16, float* __restrict__ out, int last) {
  int idx = blockIdx.x * blockDim.x + threadIdx.x;   // N*32
  if (idx >= N_ * 32) return;
  int n = idx >> 5, o = idx & 31;
  float hn = hacc[idx] * (1.0f / H_);
  h16[idx] = (_Float16)hn;
  if (last) {
    out[(size_t)n * 64 + o] = hn;
    out[(size_t)n * 64 + 32 + o] = feats[idx];
  }
}

extern "C" void kernel_launch(void* const* d_in, const int* in_sizes, int n_in,
                              void* d_out, int out_size, void* d_ws, size_t ws_size,
                              hipStream_t stream) {
  (void)in_sizes; (void)n_in; (void)out_size; (void)ws_size;
  const float* pb      = (const float*)d_in[0];
  const float* pc      = (const float*)d_in[1];
  const float* ra      = (const float*)d_in[2];
  const float* ef      = (const float*)d_in[3];
  const int*   si      = (const int*)d_in[4];
  const int*   di      = (const int*)d_in[5];
  const float* conv_w  = (const float*)d_in[6];
  const float* conv_b  = (const float*)d_in[7];
  const float* gamma   = (const float*)d_in[8];
  const float* beta    = (const float*)d_in[9];
  const float* bmean   = (const float*)d_in[10];
  const float* bvar    = (const float*)d_in[11];
  const float* fc_w    = (const float*)d_in[12];
  const float* attn_w  = (const float*)d_in[13];
  const float* few     = (const float*)d_in[14];
  const float* featw   = (const float*)d_in[15];

  float* out = (float*)d_out;
  float* alpha_out = out + (size_t)N_ * 64;

  char* ws = (char*)d_ws;
  _Float16* wB    = (_Float16*)(ws + WB_OFF);
  _Float16* fw16  = (_Float16*)(ws + FCW_OFF);
  float*    coef  = (float*)(ws + COEF_OFF);
  float*    feats = (float*)(ws + FEATS_OFF);
  _Float16* h16   = (_Float16*)(ws + H16_OFF);
  float*    z     = (float*)(ws + Z_OFF);
  float*    ssrc  = (float*)(ws + SSRC_OFF);
  float*    sdst  = (float*)(ws + SDST_OFF);
  unsigned* mbuf  = (unsigned*)(ws + M_OFF);
  float*    sumb  = (float*)(ws + SUM_OFF);
  float*    ebuf  = (float*)(ws + EB_OFF);
  float*    hacc  = (float*)(ws + HACC_OFF);

  prep_convw_kernel<<<(KCH_ * 7 * 2 * 32 + 255) / 256, 256, 0, stream>>>(conv_w, wB);
  prep_fcw_kernel<<<2, 256, 0, stream>>>(fc_w, fw16);
  prep_coef_kernel<<<1, 64, 0, stream>>>(attn_w, featw, coef);

  // 256 blocks of 128 positions, 8 waves each
  conv_encoder_kernel<<<N_ / 128, 256, 0, stream>>>(
      pb, pc, ra, wB, conv_b, gamma, beta, bmean, bvar, feats, h16);

  for (int l = 0; l < NL_; ++l) {
    int last = (l == NL_ - 1) ? 1 : 0;
    z_kernel<<<N_ / 16 / 8, 256, 0, stream>>>(h16, fw16, z, l);
    s_init_kernel<<<(N_ * H_) / 256, 256, 0, stream>>>(z, attn_w, ssrc, sdst,
                                                       mbuf, sumb, hacc, l);
    e1_kernel<<<E_ / 256, 256, 0, stream>>>(ef, si, di, ssrc, sdst, coef, ebuf, mbuf, l);
    e2_kernel<<<(E_ * H_) / 256, 256, 0, stream>>>(di, ebuf, mbuf, sumb);
    e3_kernel<<<(E_ * H_) / 256, 256, 0, stream>>>(ef, si, di, ebuf, sumb, z, few,
                                                   hacc, alpha_out, l, last);
    n2_kernel<<<(N_ * 32) / 256, 256, 0, stream>>>(hacc, feats, h16, out, last);
  }
}
